// DreamAttention_70385924046919
// MI455X (gfx1250) — compile-verified, matrix operand order checked by reading, not run on hardware
//
#include <hip/hip_runtime.h>

// ---------------------------------------------------------------------------
// DreamAttention for MI455X (gfx1250): bf16 WMMA pipeline with async-to-LDS
//   hidden @ {Wq,Wk,Wv} -> RoPE -> causal GQA flash-attention -> @ Wo
// Matmuls: v_wmma_f32_16x16x32_bf16 (fp32 accumulate).
// Tile staging: global_load_async_to_lds_b128 + s_wait_asynccnt, double-buffered.
// ---------------------------------------------------------------------------

typedef __attribute__((ext_vector_type(16))) __bf16 v16bf;
typedef __attribute__((ext_vector_type(8)))  __bf16 bf16x8;
typedef __attribute__((ext_vector_type(4)))  __bf16 bf16x4;
typedef __attribute__((ext_vector_type(8)))  float  v8f;
typedef __attribute__((ext_vector_type(4)))  float  f32x4;

#define BSZ 2
#define SEQ 2048
#define DIM 3584
#define NH  28
#define NKVH 4
#define HDM 128
#define GQA 7           // NH / NKVH

// Async copy of one 16B chunk: global -> LDS (per-lane LDS dest address).
__device__ __forceinline__ void async_b128(const __bf16* gptr, __bf16* lptr) {
  unsigned lds = (unsigned)(uintptr_t)lptr;          // low 32 bits = LDS addr
  unsigned long long ga = (unsigned long long)(uintptr_t)gptr;
  asm volatile("global_load_async_to_lds_b128 %0, %1, off"
               :: "v"(lds), "v"(ga) : "memory");
}
__device__ __forceinline__ void wait_async(int n) {
  if (n == 0) asm volatile("s_wait_asynccnt 0" ::: "memory");
  else        asm volatile("s_wait_asynccnt 8" ::: "memory");
}

// ---------------------------------------------------------------------------
// fp32 -> bf16 conversion (vectorized x4)
// ---------------------------------------------------------------------------
__global__ void cvt_f32_bf16_x4(const f32x4* __restrict__ src,
                                bf16x4* __restrict__ dst, int n4) {
  int i = blockIdx.x * blockDim.x + threadIdx.x;
  if (i < n4) {
    f32x4 x = src[i];
    bf16x4 y;
#pragma unroll
    for (int e = 0; e < 4; ++e) y[e] = (__bf16)x[e];
    dst[i] = y;
  }
}

// ---------------------------------------------------------------------------
// fp32 [K][N] -> bf16 transposed [N][K] (LDS-tiled 32x32, coalesced both ways)
// ---------------------------------------------------------------------------
__global__ __launch_bounds__(256)
void cvt_transpose(const float* __restrict__ src, __bf16* __restrict__ dst,
                   int K, int N) {
  __shared__ float tile[32][33];
  int tx = threadIdx.x, ty = threadIdx.y;          // (32, 8)
  int k0 = blockIdx.x * 32, n0 = blockIdx.y * 32;
#pragma unroll
  for (int i = 0; i < 4; ++i)
    tile[ty + i * 8][tx] = src[(size_t)(k0 + ty + i * 8) * N + n0 + tx];
  __syncthreads();
#pragma unroll
  for (int i = 0; i < 4; ++i)
    dst[(size_t)(n0 + ty + i * 8) * K + k0 + tx] =
        (__bf16)tile[tx][ty + i * 8];
}

// ---------------------------------------------------------------------------
// 16x32 bf16 fragment load (A layout; mirrored B layout on transposed store).
//   lanes 0-15 : row = lane,    elems 0-7 = K 0..7,  elems 8-15 = K 16..23
//   lanes16-31 : row = lane-16, elems 0-7 = K 8..15, elems 8-15 = K 24..31
// ---------------------------------------------------------------------------
__device__ __forceinline__ v16bf load_frag(const __bf16* base, int stride) {
  int lane = threadIdx.x & 31;
  const __bf16* p = base + (size_t)(lane & 15) * stride + ((lane >> 4) ? 8 : 0);
  bf16x8 lo = *(const bf16x8*)(p);
  bf16x8 hi = *(const bf16x8*)(p + 16);
  v16bf out;
#pragma unroll
  for (int i = 0; i < 8; ++i) { out[i] = lo[i]; out[i + 8] = hi[i]; }
  return out;
}

// ---------------------------------------------------------------------------
// Tiled bf16 GEMM:  C[M,N] = A[M,K] * W[K,N] with W supplied TRANSPOSED
// (Wt[N][K]).  Block tile 128x128 (4 waves; each wave 32 rows x 128 cols).
// Double-buffered async-to-LDS staging, BK=32.
// mode 0: fp32 store outF[row*N+col]                       (final O-proj)
// mode 1: RoPE, bf16 store [B,heads,S,HD]                  (Q, K)
// mode 2: bf16 store transposed [B,heads,HD,S]             (V)
// ---------------------------------------------------------------------------
#define BM 128
#define BN 128
#define BK 32
#define LDA 40   // 32 + 8 bf16 pad (80B rows, 16B aligned)
#define LDB 40

__device__ __forceinline__ void gemm_issue(const __bf16* __restrict__ A,
                                           const __bf16* __restrict__ Wt,
                                           int K, int tileM, int tileN, int k0,
                                           __bf16* ldsAbuf, __bf16* ldsBbuf,
                                           int tid) {
#pragma unroll
  for (int p = 0; p < 4; ++p) {                 // A: 128 rows x 32 elems
    int c = tid + p * 128;                      // chunk 0..511
    int rr = c >> 2, cc = (c & 3) * 8;
    async_b128(A + (size_t)(tileM + rr) * K + k0 + cc,
               ldsAbuf + rr * LDA + cc);
  }
#pragma unroll
  for (int p = 0; p < 4; ++p) {                 // Wt: 128 n-rows x 32 k
    int c = tid + p * 128;
    int nn = c >> 2, cc = (c & 3) * 8;
    async_b128(Wt + (size_t)(tileN + nn) * K + k0 + cc,
               ldsBbuf + nn * LDB + cc);
  }
}

__global__ __launch_bounds__(128)
void gemm_bf16(const __bf16* __restrict__ A, const __bf16* __restrict__ Wt,
               int K, int N, int mode, int nheads,
               float* __restrict__ outF, __bf16* __restrict__ outB) {
  __shared__ __bf16 ldsA[2 * BM * LDA];
  __shared__ __bf16 ldsB[2 * BN * LDB];

  const int tid  = threadIdx.x;
  const int wave = tid >> 5;
  const int lane = tid & 31;
  const int half = lane >> 4;
  const int r16  = lane & 15;
  const int tileM = blockIdx.x * BM;
  const int tileN = blockIdx.y * BN;

  v8f acc[2][8] = {};

  const int T = K / BK;
  gemm_issue(A, Wt, K, tileM, tileN, 0, ldsA, ldsB, tid);
  for (int t = 0; t < T; ++t) {
    const int cur = t & 1;
    if (t + 1 < T)
      gemm_issue(A, Wt, K, tileM, tileN, (t + 1) * BK,
                 ldsA + (cur ^ 1) * BM * LDA, ldsB + (cur ^ 1) * BN * LDB, tid);
    wait_async(t + 1 < T ? 8 : 0);
    __syncthreads();

    const __bf16* aB = ldsA + cur * BM * LDA;
    const __bf16* bB = ldsB + cur * BN * LDB;
    v16bf af[2];
#pragma unroll
    for (int m = 0; m < 2; ++m)
      af[m] = load_frag(aB + (wave * 32 + m * 16) * LDA, LDA);
#pragma unroll
    for (int nb = 0; nb < 8; ++nb) {
      v16bf bf = load_frag(bB + nb * 16 * LDB, LDB);
#pragma unroll
      for (int m = 0; m < 2; ++m)
        acc[m][nb] = __builtin_amdgcn_wmma_f32_16x16x32_bf16(
            false, af[m], false, bf, (short)0, acc[m][nb], false, false);
    }
    __syncthreads();
  }

  // ----- epilogue (C layout: lane = column, VGPR r = row r + half*8) -----
  const int head = tileN / HDM;  // BN == HD
#pragma unroll
  for (int m = 0; m < 2; ++m) {
    const int rowBase = tileM + wave * 32 + m * 16 + half * 8;
    if (mode == 0) {
#pragma unroll
      for (int nb = 0; nb < 8; ++nb) {
        int col = tileN + nb * 16 + r16;
#pragma unroll
        for (int rr = 0; rr < 8; ++rr)
          outF[(size_t)(rowBase + rr) * N + col] = acc[m][nb][rr];
      }
      continue;
    }
    if (mode == 1) {
      // RoPE: columns (d, d+64) live in acc[m][nb] / acc[m][nb+4], same lane.
#pragma unroll
      for (int nb = 0; nb < 4; ++nb) {
        int dIdx = nb * 16 + r16;  // 0..63
        float invf = __powf(1.0e6f, -(float)dIdx * (1.0f / 64.0f));
#pragma unroll
        for (int rr = 0; rr < 8; ++rr) {
          int pos = (rowBase + rr) & (SEQ - 1);
          float sn, cs;
          __sincosf((float)pos * invf, &sn, &cs);
          float lo = acc[m][nb][rr], hi = acc[m][nb + 4][rr];
          acc[m][nb][rr]     = lo * cs - hi * sn;
          acc[m][nb + 4][rr] = hi * cs + lo * sn;
        }
      }
    }
#pragma unroll
    for (int nb = 0; nb < 8; ++nb) {
      int dIdx = nb * 16 + r16;
#pragma unroll
      for (int rr = 0; rr < 8; ++rr) {
        int row = rowBase + rr;
        int b = row / SEQ, srow = row & (SEQ - 1);
        if (mode == 2)  // V stored transposed: [B, heads, HD, S]
          outB[(((size_t)b * nheads + head) * HDM + dIdx) * SEQ + srow] =
              (__bf16)acc[m][nb][rr];
        else            // Q/K: [B, heads, S, HD]
          outB[(((size_t)b * nheads + head) * SEQ + srow) * HDM + dIdx] =
              (__bf16)acc[m][nb][rr];
      }
    }
  }
}

// ---------------------------------------------------------------------------
// Flash attention: causal, GQA, scale 1/sqrt(128).
// Block = 128 threads (4 waves) covering 64 query rows of one (b, head).
// 32 keys / iteration, double-buffered async K/V tiles.
// K tile: [key][hd] from kR[B,KVH,S,HD]; V tile: [hd][key] from vT[B,KVH,HD,S].
// ---------------------------------------------------------------------------
#define LKK (HDM + 8)  // ldsK row stride
#define LVV 40         // ldsVt row stride
#define LPP 40         // ldsP row stride

__device__ __forceinline__ void attn_issue(const __bf16* kptr,
                                           const __bf16* vtptr, int kpos,
                                           __bf16* ldsKbuf, __bf16* ldsVbuf,
                                           int tid) {
#pragma unroll
  for (int p = 0; p < 4; ++p) {                 // K: 32 keys x 128 hd
    int c = tid + p * 128;                      // chunk 0..511
    int krow = c >> 4, cc = (c & 15) * 8;
    async_b128(kptr + (size_t)(kpos + krow) * HDM + cc,
               ldsKbuf + krow * LKK + cc);
  }
#pragma unroll
  for (int p = 0; p < 4; ++p) {                 // Vt: 128 hd-rows x 32 keys
    int c = tid + p * 128;
    int hd = c >> 2, cc = (c & 3) * 8;
    async_b128(vtptr + (size_t)hd * SEQ + kpos + cc,
               ldsVbuf + hd * LVV + cc);
  }
}

__global__ __launch_bounds__(128)
void attn_flash(const __bf16* __restrict__ q, const __bf16* __restrict__ k,
                const __bf16* __restrict__ vt, __bf16* __restrict__ o) {
  __shared__ __bf16 ldsK[2 * 32 * LKK];
  __shared__ __bf16 ldsVt[2 * HDM * LVV];
  __shared__ __bf16 ldsP[4 * 16 * LPP];

  const int tid  = threadIdx.x;
  const int wave = tid >> 5;
  const int lane = tid & 31;
  const int half = lane >> 4;
  const int r16  = lane & 15;

  const int qBlockBase = blockIdx.x * 64;
  const int head = blockIdx.y;
  const int b    = blockIdx.z;
  const int kvh  = head / GQA;
  const int qb   = qBlockBase + wave * 16;

  const __bf16* qptr  = q  + (((size_t)b * NH + head) * SEQ + qb) * HDM;
  const __bf16* kptr  = k  + (((size_t)b * NKVH + kvh) * SEQ) * HDM;
  const __bf16* vtptr = vt + (((size_t)b * NKVH + kvh) * HDM) * SEQ;

  v16bf qf[4];
#pragma unroll
  for (int kk = 0; kk < 4; ++kk) qf[kk] = load_frag(qptr + kk * 32, HDM);

  v8f Oacc[8] = {};
  float mstat[8], lstat[8];
#pragma unroll
  for (int rr = 0; rr < 8; ++rr) { mstat[rr] = -1e30f; lstat[rr] = 0.0f; }

  const float scale = 0.08838834764831845f;  // 1/sqrt(128)
  const int T = qBlockBase / 32 + 2;

  attn_issue(kptr, vtptr, 0, ldsK, ldsVt, tid);
  for (int t = 0; t < T; ++t) {
    const int cur  = t & 1;
    const int kpos = t * 32;
    if (t + 1 < T)
      attn_issue(kptr, vtptr, (t + 1) * 32, ldsK + (cur ^ 1) * 32 * LKK,
                 ldsVt + (cur ^ 1) * HDM * LVV, tid);
    wait_async(t + 1 < T ? 8 : 0);
    __syncthreads();

    if (kpos <= qb + 15) {  // wave-uniform causal bound (EXEC stays all-ones)
      const __bf16* kB = ldsK + cur * 32 * LKK;
      const __bf16* vB = ldsVt + cur * HDM * LVV;

      v8f sacc[2] = {};
#pragma unroll
      for (int nb = 0; nb < 2; ++nb)
#pragma unroll
        for (int kk = 0; kk < 4; ++kk) {
          v16bf bfK = load_frag(kB + nb * 16 * LKK + kk * 32, LKK);
          sacc[nb] = __builtin_amdgcn_wmma_f32_16x16x32_bf16(
              false, qf[kk], false, bfK, (short)0, sacc[nb], false, false);
        }

      float mnew[8];
#pragma unroll
      for (int rr = 0; rr < 8; ++rr) {
        int row = qb + half * 8 + rr;
        float s0 = sacc[0][rr] * scale;
        float s1 = sacc[1][rr] * scale;
        if (kpos + r16 > row)      s0 = -1e30f;
        if (kpos + 16 + r16 > row) s1 = -1e30f;
        sacc[0][rr] = s0; sacc[1][rr] = s1;
        float mx = fmaxf(s0, s1);
#pragma unroll
        for (int off = 1; off < 16; off <<= 1)
          mx = fmaxf(mx, __shfl_xor(mx, off, 32));
        mnew[rr] = fmaxf(mstat[rr], mx);
      }
#pragma unroll
      for (int rr = 0; rr < 8; ++rr) {
        float alpha = __expf(mstat[rr] - mnew[rr]);
        float p0 = __expf(sacc[0][rr] - mnew[rr]);
        float p1 = __expf(sacc[1][rr] - mnew[rr]);
        float ps = p0 + p1;
#pragma unroll
        for (int off = 1; off < 16; off <<= 1) ps += __shfl_xor(ps, off, 32);
        lstat[rr] = lstat[rr] * alpha + ps;
        mstat[rr] = mnew[rr];
#pragma unroll
        for (int nb = 0; nb < 8; ++nb) Oacc[nb][rr] *= alpha;
        int prow = half * 8 + rr;
        ldsP[(wave * 16 + prow) * LPP + r16]      = (__bf16)p0;
        ldsP[(wave * 16 + prow) * LPP + 16 + r16] = (__bf16)p1;
      }
      asm volatile("s_wait_dscnt 0" ::: "memory");  // wave-private P region
      __builtin_amdgcn_wave_barrier();

      v16bf pf = load_frag(&ldsP[wave * 16 * LPP], LPP);
#pragma unroll
      for (int nb = 0; nb < 8; ++nb) {
        v16bf vf = load_frag(vB + nb * 16 * LVV, LVV);
        Oacc[nb] = __builtin_amdgcn_wmma_f32_16x16x32_bf16(
            false, pf, false, vf, (short)0, Oacc[nb], false, false);
      }
    }
    __syncthreads();
  }

#pragma unroll
  for (int rr = 0; rr < 8; ++rr) {
    float inv = 1.0f / lstat[rr];
    int srow = qb + half * 8 + rr;
#pragma unroll
    for (int nb = 0; nb < 8; ++nb) {
      int dIdx = nb * 16 + r16;
      o[(((size_t)b * SEQ + srow) * NH + head) * HDM + dIdx] =
          (__bf16)(Oacc[nb][rr] * inv);
    }
  }
}

// ---------------------------------------------------------------------------
// Host-side launch
// ---------------------------------------------------------------------------
extern "C" void kernel_launch(void* const* d_in, const int* in_sizes, int n_in,
                              void* d_out, int out_size, void* d_ws, size_t ws_size,
                              hipStream_t stream) {
  (void)in_sizes; (void)n_in; (void)out_size; (void)ws_size;
  const float* hidden = (const float*)d_in[0];
  const float* Wq = (const float*)d_in[1];
  const float* Wk = (const float*)d_in[2];
  const float* Wv = (const float*)d_in[3];
  const float* Wo = (const float*)d_in[4];
  float* out = (float*)d_out;

  const size_t nHid = (size_t)BSZ * SEQ * DIM;
  const size_t nWq  = (size_t)DIM * NH * HDM;
  const size_t nWk  = (size_t)DIM * NKVH * HDM;
  const size_t nQ   = (size_t)BSZ * NH * SEQ * HDM;
  const size_t nK   = (size_t)BSZ * NKVH * SEQ * HDM;

  char* ws = (char*)d_ws;
  __bf16* hiddenB = (__bf16*)ws;            ws += nHid * 2;
  __bf16* WqT     = (__bf16*)ws;            ws += nWq * 2;   // [N][K]
  __bf16* WkT     = (__bf16*)ws;            ws += nWk * 2;
  __bf16* WvT     = (__bf16*)ws;            ws += nWk * 2;
  __bf16* WoT     = (__bf16*)ws;            ws += nWq * 2;
  __bf16* qR      = (__bf16*)ws;            ws += nQ * 2;    // [B,H,S,HD] roped
  __bf16* kR      = (__bf16*)ws;            ws += nK * 2;    // [B,KVH,S,HD] roped
  __bf16* vT      = (__bf16*)ws;            ws += nK * 2;    // [B,KVH,HD,S]
  __bf16* attnO   = (__bf16*)ws;                             // [B,S,H*HD]

  { // hidden: plain convert (A-matrix stays row-major)
    int n4 = (int)(nHid / 4);
    cvt_f32_bf16_x4<<<(n4 + 255) / 256, 256, 0, stream>>>(
        (const f32x4*)hidden, (bf16x4*)hiddenB, n4);
  }
  // weights: convert + transpose to [N][K]
  cvt_transpose<<<dim3(DIM / 32, (NH * HDM) / 32), dim3(32, 8), 0, stream>>>(
      Wq, WqT, DIM, NH * HDM);
  cvt_transpose<<<dim3(DIM / 32, (NKVH * HDM) / 32), dim3(32, 8), 0, stream>>>(
      Wk, WkT, DIM, NKVH * HDM);
  cvt_transpose<<<dim3(DIM / 32, (NKVH * HDM) / 32), dim3(32, 8), 0, stream>>>(
      Wv, WvT, DIM, NKVH * HDM);
  cvt_transpose<<<dim3((NH * HDM) / 32, DIM / 32), dim3(32, 8), 0, stream>>>(
      Wo, WoT, NH * HDM, DIM);

  const int Mtiles = (BSZ * SEQ) / BM;  // 32
  gemm_bf16<<<dim3(Mtiles, (NH * HDM) / BN), 128, 0, stream>>>(
      hiddenB, WqT, DIM, NH * HDM, /*mode=*/1, NH, nullptr, qR);
  gemm_bf16<<<dim3(Mtiles, (NKVH * HDM) / BN), 128, 0, stream>>>(
      hiddenB, WkT, DIM, NKVH * HDM, /*mode=*/1, NKVH, nullptr, kR);
  gemm_bf16<<<dim3(Mtiles, (NKVH * HDM) / BN), 128, 0, stream>>>(
      hiddenB, WvT, DIM, NKVH * HDM, /*mode=*/2, NKVH, nullptr, vT);

  attn_flash<<<dim3(SEQ / 64, NH, BSZ), 128, 0, stream>>>(qR, kR, vT, attnO);

  gemm_bf16<<<dim3(Mtiles, DIM / BN), 128, 0, stream>>>(
      attnO, WoT, NH * HDM, DIM, /*mode=*/0, 0, out, nullptr);
}
